// NanoGPT_53214644797454
// MI455X (gfx1250) — matrix operand order repaired and checked by reading, not compiled, not run on hardware
//
#include <hip/hip_runtime.h>
#include <hip/hip_bf16.h>
#include <math.h>

// ---------------- model constants (NanoGPT reference) ----------------
#define Bv   2
#define Lv   1024
#define Dv   1024
#define Hv   16
#define DHv  64
#define NLv  12
#define Vv   50257
#define Mv   (Bv * Lv)      // 2048 rows in the token stream
#define FFv  (4 * Dv)       // 4096

typedef unsigned short u16;

// ---------------- vector types for WMMA ----------------
typedef __attribute__((ext_vector_type(8)))  unsigned short v8u;
typedef __attribute__((ext_vector_type(16))) unsigned short v16u;
typedef __attribute__((ext_vector_type(16))) __bf16        v16bf;
typedef __attribute__((ext_vector_type(8)))  float         v8f;

__device__ __forceinline__ u16 f2bf(float f) {
  unsigned int u = __float_as_uint(f);
  u += 0x7fffu + ((u >> 16) & 1u);          // round-to-nearest-even
  return (u16)(u >> 16);
}

__device__ __forceinline__ float gelu_f(float x) {
  // jax.nn.gelu default (tanh approximation)
  const float c = 0.7978845608028654f;       // sqrt(2/pi)
  float t = tanhf(c * (x + 0.044715f * x * x * x));
  return 0.5f * x * (1.0f + t);
}

// =====================================================================
// Transpose + convert: src[K,N] f32  ->  dst[N,K] bf16
// 32x32 tiles through LDS; coalesced reads and writes.
// =====================================================================
__global__ __launch_bounds__(256) void cvt_t_kernel(
    const float* __restrict__ src, u16* __restrict__ dst, int K, int N) {
  __shared__ float tile[32][33];
  int k0 = blockIdx.y * 32;
  int n0 = blockIdx.x * 32;
  int tid = threadIdx.x;

  // read rows k0+r, cols n0+c4..+3 (coalesced along N)
  int r = tid >> 3;
  int c4 = (tid & 7) * 4;
  const float* sp = src + (size_t)(k0 + r) * N + n0 + c4;
#pragma unroll
  for (int i = 0; i < 4; ++i) {
    int n = n0 + c4 + i;
    tile[r][c4 + i] = (n < N) ? sp[i] : 0.f;
  }
  __syncthreads();

  // write col n0+cc, rows k0+r4..+3 (coalesced along K)
  int cc = tid >> 3;
  int r4 = (tid & 7) * 4;
  if (n0 + cc < N) {
    uint2 o;
    o.x = (unsigned)f2bf(tile[r4][cc])     | ((unsigned)f2bf(tile[r4 + 1][cc]) << 16);
    o.y = (unsigned)f2bf(tile[r4 + 2][cc]) | ((unsigned)f2bf(tile[r4 + 3][cc]) << 16);
    *(uint2*)(dst + (size_t)(n0 + cc) * K + k0 + r4) = o;
  }
}

// =====================================================================
// Embedding: x[row, :] = tok_emb[tokens[row], :] + pos_emb[row % L, :]
// =====================================================================
__global__ __launch_bounds__(256) void embed_kernel(
    const int* __restrict__ tokens, const float* __restrict__ tok_emb,
    const float* __restrict__ pos_emb, float* __restrict__ x) {
  int row = blockIdx.x;
  int pos = row & (Lv - 1);
  int tok = tokens[row];
  const float* te = tok_emb + (size_t)tok * Dv;
  const float* pe = pos_emb + (size_t)pos * Dv;
  float* xr = x + (size_t)row * Dv;
#pragma unroll
  for (int i = 0; i < 4; ++i) {
    int d = threadIdx.x + i * 256;
    xr[d] = te[d] + pe[d];
  }
}

// =====================================================================
// LayerNorm over D=1024, one block per row, OUTPUT IS bf16 (GEMM A-side)
// =====================================================================
__global__ __launch_bounds__(256) void ln_kernel(
    const float* __restrict__ x, const float* __restrict__ g,
    const float* __restrict__ bta, u16* __restrict__ y) {
  __shared__ float red[8];
  int row = blockIdx.x, tid = threadIdx.x;
  int w = tid >> 5, lane = tid & 31;
  int di = tid * 4;
  const float* xr = x + (size_t)row * Dv;
  float4 vv = *(const float4*)(xr + di);
  float v[4] = {vv.x, vv.y, vv.z, vv.w};

  float s = v[0] + v[1] + v[2] + v[3];
#pragma unroll
  for (int o = 16; o > 0; o >>= 1) s += __shfl_xor(s, o, 32);
  if (lane == 0) red[w] = s;
  __syncthreads();
  float tot = 0.f;
#pragma unroll
  for (int i = 0; i < 8; ++i) tot += red[i];
  float mean = tot * (1.0f / Dv);
  __syncthreads();

  float sq = 0.f;
#pragma unroll
  for (int i = 0; i < 4; ++i) { float d = v[i] - mean; sq += d * d; }
#pragma unroll
  for (int o = 16; o > 0; o >>= 1) sq += __shfl_xor(sq, o, 32);
  if (lane == 0) red[w] = sq;
  __syncthreads();
  float tot2 = 0.f;
#pragma unroll
  for (int i = 0; i < 8; ++i) tot2 += red[i];
  float rs = rsqrtf(tot2 * (1.0f / Dv) + 1e-6f);

  float4 gg = *(const float4*)(g + di);
  float4 bb = *(const float4*)(bta + di);
  float y0 = (v[0] - mean) * rs * gg.x + bb.x;
  float y1 = (v[1] - mean) * rs * gg.y + bb.y;
  float y2 = (v[2] - mean) * rs * gg.z + bb.z;
  float y3 = (v[3] - mean) * rs * gg.w + bb.w;
  uint2 o;
  o.x = (unsigned)f2bf(y0) | ((unsigned)f2bf(y1) << 16);
  o.y = (unsigned)f2bf(y2) | ((unsigned)f2bf(y3) << 16);
  *(uint2*)(y + (size_t)row * Dv + di) = o;
}

// =====================================================================
// GEMM: C[M,N] = epilogue( A[M,K](bf16) @ Wt[N,K](bf16, pre-transposed) )
//   epilogue: (+bias) -> (gelu) -> (+residual); writes f32 C and/or bf16 Cbf
// Block tile 128x128, 8 waves (4M x 2N), wave tile 32x64 = 8 accumulators.
// Double-buffered LDS; A and B staged with b128 copies (no scatter).
// RAGGED=false (N % 128 == 0): no guards anywhere -> pure global_load_b128.
// =====================================================================
#define BM   128
#define BN   128
#define KS   32
#define LSTR 40    // LDS row stride (halves): 80B rows -> 16B aligned, bank-safe

template <bool RAGGED>
__global__ __launch_bounds__(256) void gemm_bf16_kernel(
    const u16* __restrict__ A, const u16* __restrict__ Wt,
    const float* __restrict__ bias, const float* __restrict__ residual,
    float* __restrict__ C, u16* __restrict__ Cbf,
    int M, int K, int N, int do_gelu) {
  __shared__ alignas(16) u16 Alds[2][BM * LSTR];  // 2 x 10 KB
  __shared__ alignas(16) u16 Blds[2][BN * LSTR];  // 2 x 10 KB

  int tid = threadIdx.x;
  int m0 = blockIdx.y * BM;
  int n0 = blockIdx.x * BN;
  int w = tid >> 5, lane = tid & 31;
  int wm = w >> 1, wn = w & 1;          // 4 x 2 wave grid
  int r = lane & 15, hi = lane >> 4;

  const v8f vzero = {0.f, 0.f, 0.f, 0.f, 0.f, 0.f, 0.f, 0.f};
  v8f acc[2][4] = {{vzero, vzero, vzero, vzero}, {vzero, vzero, vzero, vzero}};

  // cooperative loader indices: 16 halves (32B) per thread per tile
  int a_row = tid >> 1;                 // 0..127
  int a_seg = (tid & 1) * 16;           // 0 or 16
  int b_col = tid >> 1;                 // 0..127
  int b_seg = (tid & 1) * 16;           // 0 or 16

  uint4 apre0, apre1, bpre0, bpre1;
  const uint4 z4 = {0u, 0u, 0u, 0u};

  auto loadG = [&](int k0) {
    const u16* ap = A + (size_t)(m0 + a_row) * K + k0 + a_seg;
    apre0 = *(const uint4*)ap;
    apre1 = *(const uint4*)(ap + 8);
    const u16* wp = Wt + (size_t)(n0 + b_col) * K + k0 + b_seg;
    if (!RAGGED || (n0 + b_col < N)) {
      bpre0 = *(const uint4*)wp;
      bpre1 = *(const uint4*)(wp + 8);
    } else {
      bpre0 = z4; bpre1 = z4;
    }
  };

  auto storeL = [&](int buf) {
    *(uint4*)&Alds[buf][a_row * LSTR + a_seg]     = apre0;
    *(uint4*)&Alds[buf][a_row * LSTR + a_seg + 8] = apre1;
    *(uint4*)&Blds[buf][b_col * LSTR + b_seg]     = bpre0;
    *(uint4*)&Blds[buf][b_col * LSTR + b_seg + 8] = bpre1;
  };

  auto compute = [&](int buf) {
    // gather all 6 fragments first so the 8 WMMAs can issue back-to-back
    v16bf afrag[2], bfrag[4];
#pragma unroll
    for (int mi = 0; mi < 2; ++mi) {
      int abase = (wm * 32 + mi * 16 + r) * LSTR + hi * 8;
      v8u alo = *(const v8u*)&Alds[buf][abase];
      v8u ahi = *(const v8u*)&Alds[buf][abase + 16];
      v16u afu;
#pragma unroll
      for (int i = 0; i < 8; ++i) { afu[i] = alo[i]; afu[i + 8] = ahi[i]; }
      afrag[mi] = __builtin_bit_cast(v16bf, afu);
    }
#pragma unroll
    for (int ni = 0; ni < 4; ++ni) {
      int bbase = (wn * 64 + ni * 16 + r) * LSTR + hi * 8;
      v8u blo = *(const v8u*)&Blds[buf][bbase];
      v8u bhi = *(const v8u*)&Blds[buf][bbase + 16];
      v16u bfu;
#pragma unroll
      for (int i = 0; i < 8; ++i) { bfu[i] = blo[i]; bfu[i + 8] = bhi[i]; }
      bfrag[ni] = __builtin_bit_cast(v16bf, bfu);
    }
#pragma unroll
    for (int ni = 0; ni < 4; ++ni)
#pragma unroll
      for (int mi = 0; mi < 2; ++mi)
        acc[mi][ni] = __builtin_amdgcn_wmma_f32_16x16x32_bf16(
            false, afrag[mi], false, bfrag[ni], (short)0, acc[mi][ni], false, false);
  };

  int nk = K / KS;
  loadG(0);
  storeL(0);
  __syncthreads();
  for (int t = 0; t < nk; ++t) {
    int buf = t & 1;
    if (t + 1 < nk) loadG((t + 1) * KS);   // prefetch to regs, overlap WMMA
    compute(buf);
    if (t + 1 < nk) {
      storeL(buf ^ 1);
      __syncthreads();
    }
  }

  // ---- epilogue + store
#pragma unroll
  for (int mi = 0; mi < 2; ++mi) {
#pragma unroll
    for (int ni = 0; ni < 4; ++ni) {
#pragma unroll
      for (int f = 0; f < 8; ++f) {
        int row = m0 + wm * 32 + mi * 16 + hi * 8 + f;
        int col = n0 + wn * 64 + ni * 16 + r;
        if (!RAGGED || (col < N)) {
          float val = acc[mi][ni][f];
          if (bias) val += bias[col];
          if (do_gelu) val = gelu_f(val);
          if (residual) val += residual[(size_t)row * N + col];
          if (C)   C[(size_t)row * N + col] = val;
          if (Cbf) Cbf[(size_t)row * N + col] = f2bf(val);
        }
      }
    }
  }
}

// =====================================================================
// Causal flash attention (f32 q/k/v in, bf16 out), head dim 64.
// Block = (b,h) x 64 q-rows; 8 waves x 8 rows; lanes = keys; online softmax.
// =====================================================================
#define QB 64
#define KT 32
#define VSTR 66   // padded f32 stride, conflict-free

__global__ __launch_bounds__(256) void attn_kernel(
    const float* __restrict__ q, const float* __restrict__ k,
    const float* __restrict__ v, u16* __restrict__ o) {
  __shared__ float Qs[QB * VSTR];
  __shared__ float Ks[KT * VSTR];
  __shared__ float Vs[KT * VSTR];

  int bh = blockIdx.x;
  int b = bh >> 4, h = bh & (Hv - 1);
  int q0 = blockIdx.y * QB;
  int tid = threadIdx.x, w = tid >> 5, lane = tid & 31;

  {
    int row = tid >> 2, ds = (tid & 3) * 16;
    const float* qp = q + (size_t)(b * Lv + q0 + row) * Dv + h * DHv + ds;
#pragma unroll
    for (int i = 0; i < 16; ++i) Qs[row * VSTR + ds + i] = qp[i];
  }
  __syncthreads();

  float m[8], l[8], a0[8], a1[8];
#pragma unroll
  for (int rr = 0; rr < 8; ++rr) { m[rr] = -1e30f; l[rr] = 0.f; a0[rr] = 0.f; a1[rr] = 0.f; }

  int rbase = w * 8;
  int kv_row = tid >> 3, kv_ds = (tid & 7) * 8;

  for (int jt = 0; jt <= q0 + QB - 1; jt += KT) {
    __syncthreads();
    {
      size_t base = (size_t)(b * Lv + jt + kv_row) * Dv + h * DHv + kv_ds;
#pragma unroll
      for (int i = 0; i < 8; ++i) {
        Ks[kv_row * VSTR + kv_ds + i] = k[base + i];
        Vs[kv_row * VSTR + kv_ds + i] = v[base + i];
      }
    }
    __syncthreads();

    for (int rr = 0; rr < 8; ++rr) {
      int qrow = q0 + rbase + rr;
      const float* qp = &Qs[(rbase + rr) * VSTR];
      const float* kp = &Ks[lane * VSTR];
      float s = 0.f;
#pragma unroll
      for (int d = 0; d < DHv; ++d) s += qp[d] * kp[d];
      s *= 0.125f;                                  // 1/sqrt(64)
      if (jt + lane > qrow) s = -1e30f;             // causal mask

      float tmax = s;
#pragma unroll
      for (int off = 16; off > 0; off >>= 1) tmax = fmaxf(tmax, __shfl_xor(tmax, off, 32));
      float mnew = fmaxf(m[rr], tmax);
      float corr = __expf(m[rr] - mnew);
      float p = __expf(s - mnew);
      float psum = p;
#pragma unroll
      for (int off = 16; off > 0; off >>= 1) psum += __shfl_xor(psum, off, 32);
      l[rr] = l[rr] * corr + psum;
      a0[rr] *= corr; a1[rr] *= corr;
#pragma unroll
      for (int j = 0; j < KT; ++j) {
        float pj = __shfl(p, j, 32);
        a0[rr] += pj * Vs[j * VSTR + lane * 2];
        a1[rr] += pj * Vs[j * VSTR + lane * 2 + 1];
      }
      m[rr] = mnew;
    }
  }

#pragma unroll
  for (int rr = 0; rr < 8; ++rr) {
    int qrow = q0 + rbase + rr;
    float inv = 1.0f / l[rr];
    size_t base = (size_t)(b * Lv + qrow) * Dv + h * DHv + lane * 2;
    unsigned int pk = (unsigned)f2bf(a0[rr] * inv) | ((unsigned)f2bf(a1[rr] * inv) << 16);
    *(unsigned int*)(o + base) = pk;
  }
}

// =====================================================================
// Host-side launcher
// =====================================================================
extern "C" void kernel_launch(void* const* d_in, const int* in_sizes, int n_in,
                              void* d_out, int out_size, void* d_ws, size_t ws_size,
                              hipStream_t stream) {
  (void)in_sizes; (void)n_in; (void)out_size; (void)ws_size;

  const int*   tokens  = (const int*)  d_in[0];
  const float* tok_emb = (const float*)d_in[1];
  const float* pos_emb = (const float*)d_in[2];
  const float* ln1_s   = (const float*)d_in[3];
  const float* ln1_b   = (const float*)d_in[4];
  const float* Wq      = (const float*)d_in[5];
  const float* Wk      = (const float*)d_in[6];
  const float* Wv      = (const float*)d_in[7];
  const float* Wo      = (const float*)d_in[8];
  const float* ln2_s   = (const float*)d_in[9];
  const float* ln2_b   = (const float*)d_in[10];
  const float* W1      = (const float*)d_in[11];
  const float* b1      = (const float*)d_in[12];
  const float* W2      = (const float*)d_in[13];
  const float* b2      = (const float*)d_in[14];
  const float* lnf_s   = (const float*)d_in[15];
  const float* lnf_b   = (const float*)d_in[16];
  const float* Whead   = (const float*)d_in[17];
  const float* bhead   = (const float*)d_in[18];
  float* out = (float*)d_out;

  // ---- workspace carving (byte based)
  const size_t MD = (size_t)Mv * Dv;
  char* p = (char*)d_ws;
  float* x   = (float*)p; p += MD * 4;            // residual stream (f32)
  float* qb  = (float*)p; p += MD * 4;
  float* kb  = (float*)p; p += MD * 4;
  float* vb  = (float*)p; p += MD * 4;
  u16* hbf   = (u16*)p;   p += MD * 2;            // LN output (bf16)
  u16* ab    = (u16*)p;   p += MD * 2;            // attention output (bf16)
  u16* mhbf  = (u16*)p;   p += (size_t)Mv * FFv * 2;  // MLP hidden (bf16)
  u16* wbf   = (u16*)p;   p += (size_t)Dv * Vv * 2;   // bf16 W^T staging

  dim3 blk(256);
  dim3 gD (Dv  / BN, Mv / BM);                    // (8, 16)
  dim3 gFF(FFv / BN, Mv / BM);                    // (32, 16)
  dim3 gV ((Vv + BN - 1) / BN, Mv / BM);          // (393, 16)
  dim3 gAttn(Bv * Hv, Lv / QB);

  // transpose-convert grids: x = ceil(N/32), y = K/32
  dim3 tDD(Dv / 32, Dv / 32);                     // D x D weights
  dim3 tDF(FFv / 32, Dv / 32);                    // W1: K=D,  N=FF
  dim3 tFD(Dv / 32, FFv / 32);                    // W2: K=FF, N=D
  dim3 tDV((Vv + 31) / 32, Dv / 32);              // head: K=D, N=V

  embed_kernel<<<Mv, blk, 0, stream>>>(tokens, tok_emb, pos_emb, x);

  for (int layer = 0; layer < NLv; ++layer) {
    const size_t DD = (size_t)Dv * Dv;
    const float* Wq_l = Wq + layer * DD;
    const float* Wk_l = Wk + layer * DD;
    const float* Wv_l = Wv + layer * DD;
    const float* Wo_l = Wo + layer * DD;
    const float* W1_l = W1 + layer * (size_t)Dv * FFv;
    const float* W2_l = W2 + layer * (size_t)FFv * Dv;
    const float* b1_l = b1 + layer * (size_t)FFv;
    const float* b2_l = b2 + layer * (size_t)Dv;

    ln_kernel<<<Mv, blk, 0, stream>>>(x, ln1_s + layer * Dv, ln1_b + layer * Dv, hbf);

    cvt_t_kernel<<<tDD, blk, 0, stream>>>(Wq_l, wbf, Dv, Dv);
    gemm_bf16_kernel<false><<<gD, blk, 0, stream>>>(hbf, wbf, nullptr, nullptr, qb, nullptr, Mv, Dv, Dv, 0);
    cvt_t_kernel<<<tDD, blk, 0, stream>>>(Wk_l, wbf, Dv, Dv);
    gemm_bf16_kernel<false><<<gD, blk, 0, stream>>>(hbf, wbf, nullptr, nullptr, kb, nullptr, Mv, Dv, Dv, 0);
    cvt_t_kernel<<<tDD, blk, 0, stream>>>(Wv_l, wbf, Dv, Dv);
    gemm_bf16_kernel<false><<<gD, blk, 0, stream>>>(hbf, wbf, nullptr, nullptr, vb, nullptr, Mv, Dv, Dv, 0);

    attn_kernel<<<gAttn, blk, 0, stream>>>(qb, kb, vb, ab);

    // x = x + attn_out @ Wo   (in-place safe: each element read once then written)
    cvt_t_kernel<<<tDD, blk, 0, stream>>>(Wo_l, wbf, Dv, Dv);
    gemm_bf16_kernel<false><<<gD, blk, 0, stream>>>(ab, wbf, nullptr, x, x, nullptr, Mv, Dv, Dv, 0);

    ln_kernel<<<Mv, blk, 0, stream>>>(x, ln2_s + layer * Dv, ln2_b + layer * Dv, hbf);

    cvt_t_kernel<<<tDF, blk, 0, stream>>>(W1_l, wbf, Dv, FFv);
    gemm_bf16_kernel<false><<<gFF, blk, 0, stream>>>(hbf, wbf, b1_l, nullptr, nullptr, mhbf, Mv, Dv, FFv, 1);
    cvt_t_kernel<<<tFD, blk, 0, stream>>>(W2_l, wbf, FFv, Dv);
    gemm_bf16_kernel<false><<<gD, blk, 0, stream>>>(mhbf, wbf, b2_l, x, x, nullptr, Mv, FFv, Dv, 0);
  }

  ln_kernel<<<Mv, blk, 0, stream>>>(x, lnf_s, lnf_b, hbf);
  cvt_t_kernel<<<tDV, blk, 0, stream>>>(Whead, wbf, Dv, Vv);
  gemm_bf16_kernel<true><<<gV, blk, 0, stream>>>(hbf, wbf, bhead, nullptr, out, nullptr, Mv, Dv, Vv, 0);
}
